// BiT_Phoneme_8323646620217
// MI455X (gfx1250) — compile-verified, matrix operand order checked by reading, not run on hardware
//
#include <hip/hip_runtime.h>
#include <math.h>

// ---------------- types for CDNA5 WMMA ----------------
typedef __attribute__((ext_vector_type(16))) __bf16 v16bf;
typedef __attribute__((ext_vector_type(8)))  __bf16 v8bf;
typedef __attribute__((ext_vector_type(8)))  float  v8f;

// ---------------- model constants ----------------
#define BATCH   4
#define TLEN    2048
#define PWF     256
#define NPATCH  512            // TLEN / 4
#define ROWS    2048           // BATCH * NPATCH
#define DIMC    1024
#define HEADS   16
#define DHEAD   64
#define INNERC  1024
#define MLPC    4096
#define NCLS1   41
#define KREL    399            // 2*200-1

// =====================================================
// f32 -> bf16 weight conversion (grid-stride)
// =====================================================
__global__ void f32_to_bf16_kernel(const float* __restrict__ src,
                                   __bf16* __restrict__ dst, int n) {
  int i = blockIdx.x * blockDim.x + threadIdx.x;
  int stride = gridDim.x * blockDim.x;
  for (; i < n; i += stride) dst[i] = (__bf16)src[i];
}

// =====================================================
// Gaussian depthwise conv (k=20, pads 9/10) + patchify + LN1 -> bf16
// one block per patch row (B*N = 2048), 256 threads = feature id
// =====================================================
__global__ __launch_bounds__(256)
void patchify_ln_kernel(const float* __restrict__ in,
                        const float* __restrict__ g,
                        const float* __restrict__ bb,
                        __bf16* __restrict__ h) {
  const int bn = blockIdx.x;
  const int b  = bn >> 9;
  const int n  = bn & 511;
  const int f  = threadIdx.x;

  // build normalized gaussian kernel (prefactor cancels after normalization)
  float gk[20];
  float gs = 0.0f;
#pragma unroll
  for (int u = 0; u < 20; ++u) {
    float d = ((float)u - 9.5f) * 0.5f;   // (u - mean)/sigma, sigma=2
    gk[u] = expf(-0.5f * d * d);
    gs += gk[u];
  }
  const float ginv = 1.0f / gs;

  float v[4];
#pragma unroll
  for (int p = 0; p < 4; ++p) {
    const int t = n * 4 + p;
    float acc = 0.0f;
#pragma unroll
    for (int u = 0; u < 20; ++u) {
      const int tt = t + u - 9;
      if (tt >= 0 && tt < TLEN)
        acc += gk[u] * in[((size_t)b * TLEN + tt) * PWF + f];
    }
    v[p] = acc * ginv;
  }

  // LayerNorm over the 1024-wide patch
  __shared__ float red[256];
  float part = v[0] + v[1] + v[2] + v[3];
  red[f] = part; __syncthreads();
  for (int st = 128; st > 0; st >>= 1) {
    if (f < st) red[f] += red[f + st];
    __syncthreads();
  }
  const float mean = red[0] * (1.0f / 1024.0f);
  __syncthreads();
  float p2 = 0.0f;
#pragma unroll
  for (int p = 0; p < 4; ++p) { float d = v[p] - mean; p2 += d * d; }
  red[f] = p2; __syncthreads();
  for (int st = 128; st > 0; st >>= 1) {
    if (f < st) red[f] += red[f + st];
    __syncthreads();
  }
  const float inv = rsqrtf(red[0] * (1.0f / 1024.0f) + 1e-5f);

#pragma unroll
  for (int p = 0; p < 4; ++p) {
    const int e = p * 256 + f;
    const float y = (v[p] - mean) * inv * g[e] + bb[e];
    h[(size_t)bn * DIMC + e] = (__bf16)y;
  }
}

// =====================================================
// Row LayerNorm (C = 1024): one block per row, optional f32/bf16 outputs
// =====================================================
__global__ __launch_bounds__(256)
void ln_kernel(const float* __restrict__ x,
               const float* __restrict__ g,
               const float* __restrict__ bb,
               float* __restrict__ outF,
               __bf16* __restrict__ outB) {
  const int row = blockIdx.x;
  const int t = threadIdx.x;
  const float* xr = x + (size_t)row * DIMC;
  float v[4];
#pragma unroll
  for (int k = 0; k < 4; ++k) v[k] = xr[t + k * 256];

  __shared__ float red[256];
  red[t] = v[0] + v[1] + v[2] + v[3]; __syncthreads();
  for (int st = 128; st > 0; st >>= 1) {
    if (t < st) red[t] += red[t + st];
    __syncthreads();
  }
  const float mean = red[0] * (1.0f / 1024.0f);
  __syncthreads();
  float p2 = 0.0f;
#pragma unroll
  for (int k = 0; k < 4; ++k) { float d = v[k] - mean; p2 += d * d; }
  red[t] = p2; __syncthreads();
  for (int st = 128; st > 0; st >>= 1) {
    if (t < st) red[t] += red[t + st];
    __syncthreads();
  }
  const float inv = rsqrtf(red[0] * (1.0f / 1024.0f) + 1e-5f);

#pragma unroll
  for (int k = 0; k < 4; ++k) {
    const int e = t + k * 256;
    const float y = (v[k] - mean) * inv * g[e] + bb[e];
    const size_t idx = (size_t)row * DIMC + e;
    if (outF) outF[idx] = y;
    if (outB) outB[idx] = (__bf16)y;
  }
}

// =====================================================
// WMMA bf16 GEMM:  C[M,N] = A[M,K] * W[N,K]^T  (+bias)(+gelu)(+residual)
// block = 256 threads (8 waves), block tile 128(M) x 128(N),
// wave grid 2(M) x 4(N), wave tile 64x32 -> 8 WMMAs per K-chunk per wave
// (6 fragment loads : 8 wmma).  K chunks of 32.
// LDS pitch 40 halves (80B = 20 banks) -> conflict-free b128 frag reads.
// =====================================================
__device__ __forceinline__ v16bf load_frag(const __bf16* lds, int row_base, int lane) {
  const int r  = row_base + (lane & 15);
  const int c0 = (lane < 16) ? 0 : 8;
  union { v16bf v; v8bf h[2]; } f;
  f.h[0] = *(const v8bf*)(lds + r * 40 + c0);
  f.h[1] = *(const v8bf*)(lds + r * 40 + c0 + 16);
  return f.v;
}

__global__ __launch_bounds__(256)
void gemm_bf16_kernel(const __bf16* __restrict__ A,
                      const __bf16* __restrict__ W,
                      const float* __restrict__ bias,
                      const float* __restrict__ residual,
                      float* __restrict__ outF,
                      __bf16* __restrict__ outB,
                      int M, int N, int K, int do_gelu) {
  __shared__ __align__(16) __bf16 As[128 * 40];
  __shared__ __align__(16) __bf16 Bs[128 * 40];

  const int tid  = threadIdx.x;
  const int lane = tid & 31;
  const int wave = tid >> 5;
  const int wm   = wave >> 2;   // 0..1  (64 rows each)
  const int wn   = wave & 3;    // 0..3  (32 cols each)
  const int row0 = blockIdx.y * 128;
  const int col0 = blockIdx.x * 128;

  v8f acc[4][2];
#pragma unroll
  for (int ma = 0; ma < 4; ++ma) {
    acc[ma][0] = {};
    acc[ma][1] = {};
  }

  for (int kc = 0; kc < K; kc += 32) {
    // stage A tile: 128 rows x 32 halves; 512 chunks of 8 halves (16B)
#pragma unroll
    for (int s = 0; s < 2; ++s) {
      const int c = tid + s * 256;
      const int row = c >> 2, part = c & 3;
      *(v8bf*)&As[row * 40 + part * 8] =
          *(const v8bf*)(A + (size_t)(row0 + row) * K + kc + part * 8);
    }
    // stage B tile: 128 rows x 32 halves; 512 chunks
#pragma unroll
    for (int s = 0; s < 2; ++s) {
      const int c = tid + s * 256;
      const int row = c >> 2, part = c & 3;
      *(v8bf*)&Bs[row * 40 + part * 8] =
          *(const v8bf*)(W + (size_t)(col0 + row) * K + kc + part * 8);
    }
    __syncthreads();

    const v16bf b0 = load_frag(Bs, wn * 32,      lane);
    const v16bf b1 = load_frag(Bs, wn * 32 + 16, lane);
    v16bf a[4];
#pragma unroll
    for (int ma = 0; ma < 4; ++ma) a[ma] = load_frag(As, wm * 64 + ma * 16, lane);

#pragma unroll
    for (int ma = 0; ma < 4; ++ma) {
      acc[ma][0] = __builtin_amdgcn_wmma_f32_16x16x32_bf16(false, a[ma], false, b0,
                   (short)0, acc[ma][0], false, false);
      acc[ma][1] = __builtin_amdgcn_wmma_f32_16x16x32_bf16(false, a[ma], false, b1,
                   (short)0, acc[ma][1], false, false);
    }
    __syncthreads();
  }

  // epilogue: C/D layout -> lane holds col (lane&15), rows i / i+8
#pragma unroll
  for (int ma = 0; ma < 4; ++ma) {
#pragma unroll
    for (int nb = 0; nb < 2; ++nb) {
      const int n_g = col0 + wn * 32 + nb * 16 + (lane & 15);
      const float bia = bias ? bias[n_g] : 0.0f;
#pragma unroll
      for (int i = 0; i < 8; ++i) {
        const int m_g = row0 + wm * 64 + ma * 16 + ((lane < 16) ? i : i + 8);
        float v = acc[ma][nb][i] + bia;
        if (do_gelu) v = 0.5f * v * (1.0f + erff(v * 0.7071067811865475f));
        const size_t idx = (size_t)m_g * N + n_g;
        if (residual) v += residual[idx];
        if (outF) outF[idx] = v;
        if (outB) outB[idx] = (__bf16)v;
      }
    }
  }
}

// =====================================================
// Causal attention with relative-position bias, online softmax.
// one block per (batch, head); thread = query row i (512 threads).
// K/V tiles of 16 rows staged in LDS; per-tile max merge.
// =====================================================
__global__ __launch_bounds__(512)
void attn_kernel(const __bf16* __restrict__ qkv,
                 const float* __restrict__ rel,
                 __bf16* __restrict__ out) {
  const int bh = blockIdx.x;
  const int b  = bh >> 4;
  const int hh = bh & 15;
  const int i  = threadIdx.x;   // query row 0..511

  __shared__ float kt[16 * 64];
  __shared__ float vt[16 * 64];
  __shared__ float relb[KREL];
  for (int u = i; u < KREL; u += 512) relb[u] = rel[u];

  float q[64];
  {
    const size_t qb = ((size_t)(b * NPATCH + i)) * 3072 + hh * 64;
#pragma unroll
    for (int d = 0; d < 64; ++d) q[d] = (float)qkv[qb + d] * 0.125f; // DH^-0.5
  }
  float o[64];
#pragma unroll
  for (int d = 0; d < 64; ++d) o[d] = 0.0f;
  float mrun = -1e30f, lrun = 0.0f;

  for (int jt = 0; jt < 32; ++jt) {
    const int j0 = jt * 16;
    // stage 16 K rows and 16 V rows
    for (int e = i; e < 1024; e += 512) {
      const int jl = e >> 6, d = e & 63;
      const size_t kb = ((size_t)(b * NPATCH + j0 + jl)) * 3072 + hh * 64 + d;
      kt[e] = (float)qkv[kb + 1024];
      vt[e] = (float)qkv[kb + 2048];
    }
    __syncthreads();

    if (j0 <= i) {
      float sv[16];
#pragma unroll
      for (int jl = 0; jl < 16; ++jl) {
        const int j = j0 + jl;
        if (j <= i) {
          float s = 0.0f;
#pragma unroll
          for (int d = 0; d < 64; ++d) s += q[d] * kt[jl * 64 + d];
          int dd = i - j; if (dd > 199) dd = 199;
          sv[jl] = s + relb[dd + 199];
        } else {
          sv[jl] = -1e30f;
        }
      }
      float tmax = sv[0];
#pragma unroll
      for (int jl = 1; jl < 16; ++jl) tmax = fmaxf(tmax, sv[jl]);
      const float mnew = fmaxf(mrun, tmax);
      const float corr = expf(mrun - mnew);
      float psum = 0.0f;
#pragma unroll
      for (int jl = 0; jl < 16; ++jl) {
        const float p = (sv[jl] > -1e29f) ? expf(sv[jl] - mnew) : 0.0f;
        sv[jl] = p;
        psum += p;
      }
      lrun = lrun * corr + psum;
#pragma unroll
      for (int d = 0; d < 64; ++d) {
        float a = 0.0f;
#pragma unroll
        for (int jl = 0; jl < 16; ++jl) a += sv[jl] * vt[jl * 64 + d];
        o[d] = o[d] * corr + a;
      }
      mrun = mnew;
    }
    __syncthreads();
  }

  const float inv = 1.0f / lrun;
  const size_t ob = ((size_t)(b * NPATCH + i)) * INNERC + hh * 64;
#pragma unroll
  for (int d = 0; d < 64; ++d) out[ob + d] = (__bf16)(o[d] * inv);
}

// =====================================================
// Final projection: out[row, 0..40] = ln(x) @ proj_w.T + proj_b
// one block per row; x row staged in LDS.
// =====================================================
__global__ __launch_bounds__(64)
void proj_kernel(const __bf16* __restrict__ h,
                 const float* __restrict__ w,
                 const float* __restrict__ bb,
                 float* __restrict__ out) {
  const int row = blockIdx.x;
  const int t = threadIdx.x;
  __shared__ float xs[DIMC];
  for (int e = t; e < DIMC; e += 64) xs[e] = (float)h[(size_t)row * DIMC + e];
  __syncthreads();
  if (t < NCLS1) {
    float a = bb[t];
    const float* wr = w + (size_t)t * DIMC;
    for (int k = 0; k < DIMC; ++k) a += xs[k] * wr[k];
    out[(size_t)row * NCLS1 + t] = a;
  }
}

// =====================================================
// host launcher
// =====================================================
extern "C" void kernel_launch(void* const* d_in, const int* in_sizes, int n_in,
                              void* d_out, int out_size, void* d_ws, size_t ws_size,
                              hipStream_t stream) {
  const float* neuralInput  = (const float*)d_in[0];
  const float* patch_ln1_g  = (const float*)d_in[1];
  const float* patch_ln1_b  = (const float*)d_in[2];
  const float* patch_w      = (const float*)d_in[3];
  const float* patch_b      = (const float*)d_in[4];
  const float* patch_ln2_g  = (const float*)d_in[5];
  const float* patch_ln2_b  = (const float*)d_in[6];
  const float* attn_ln_g    = (const float*)d_in[7];
  const float* attn_ln_b    = (const float*)d_in[8];
  const float* qkv_w        = (const float*)d_in[9];
  const float* out_w        = (const float*)d_in[10];
  const float* out_b        = (const float*)d_in[11];
  const float* rel_tab      = (const float*)d_in[12];
  const float* ffn_ln_g     = (const float*)d_in[13];
  const float* ffn_ln_b     = (const float*)d_in[14];
  const float* ffn_w1       = (const float*)d_in[15];
  const float* ffn_b1       = (const float*)d_in[16];
  const float* ffn_w2       = (const float*)d_in[17];
  const float* ffn_b2       = (const float*)d_in[18];
  const float* final_ln_g   = (const float*)d_in[19];
  const float* final_ln_b   = (const float*)d_in[20];
  const float* proj_w       = (const float*)d_in[21];
  const float* proj_b       = (const float*)d_in[22];
  // d_in[23] = X_len : unused by the reference computation

  // workspace carve-up (~71 MB)
  char* p = (char*)d_ws;
  float*  x_f32 = (float*)p;  p += (size_t)ROWS * DIMC * 4;
  __bf16* h_b   = (__bf16*)p; p += (size_t)ROWS * DIMC * 2;
  __bf16* qkv_b = (__bf16*)p; p += (size_t)ROWS * 3072 * 2;
  __bf16* ao_b  = (__bf16*)p; p += (size_t)ROWS * INNERC * 2;
  __bf16* mlp_b = (__bf16*)p; p += (size_t)ROWS * MLPC * 2;
  __bf16* wq    = (__bf16*)p; p += (size_t)3072 * DIMC * 2;
  __bf16* wo    = (__bf16*)p; p += (size_t)DIMC * DIMC * 2;
  __bf16* w1    = (__bf16*)p; p += (size_t)MLPC * DIMC * 2;
  __bf16* w2    = (__bf16*)p; p += (size_t)DIMC * MLPC * 2;

  // --- front end: smooth + patchify + LN1 -> h_b (bf16) ---
  patchify_ln_kernel<<<ROWS, 256, 0, stream>>>(neuralInput, patch_ln1_g,
                                               patch_ln1_b, h_b);
  // patch embed GEMM (weights staged in wo), then LN2 -> residual stream x_f32
  f32_to_bf16_kernel<<<1024, 256, 0, stream>>>(patch_w, wo, DIMC * DIMC);
  gemm_bf16_kernel<<<dim3(DIMC / 128, ROWS / 128), 256, 0, stream>>>(
      h_b, wo, patch_b, nullptr, x_f32, nullptr, ROWS, DIMC, DIMC, 0);
  ln_kernel<<<ROWS, 256, 0, stream>>>(x_f32, patch_ln2_g, patch_ln2_b,
                                      x_f32, nullptr);

  for (int l = 0; l < 12; ++l) {
    f32_to_bf16_kernel<<<2048, 256, 0, stream>>>(qkv_w + (size_t)l * 3072 * DIMC,
                                                 wq, 3072 * DIMC);
    f32_to_bf16_kernel<<<1024, 256, 0, stream>>>(out_w + (size_t)l * DIMC * DIMC,
                                                 wo, DIMC * DIMC);
    f32_to_bf16_kernel<<<2048, 256, 0, stream>>>(ffn_w1 + (size_t)l * MLPC * DIMC,
                                                 w1, MLPC * DIMC);
    f32_to_bf16_kernel<<<2048, 256, 0, stream>>>(ffn_w2 + (size_t)l * DIMC * MLPC,
                                                 w2, DIMC * MLPC);

    // attention block
    ln_kernel<<<ROWS, 256, 0, stream>>>(x_f32, attn_ln_g + l * DIMC,
                                        attn_ln_b + l * DIMC, nullptr, h_b);
    gemm_bf16_kernel<<<dim3(3072 / 128, ROWS / 128), 256, 0, stream>>>(
        h_b, wq, nullptr, nullptr, nullptr, qkv_b, ROWS, 3072, DIMC, 0);
    attn_kernel<<<BATCH * HEADS, 512, 0, stream>>>(qkv_b, rel_tab + l * KREL,
                                                   ao_b);
    gemm_bf16_kernel<<<dim3(DIMC / 128, ROWS / 128), 256, 0, stream>>>(
        ao_b, wo, out_b + l * DIMC, x_f32, x_f32, nullptr, ROWS, DIMC, INNERC, 0);

    // FFN block
    ln_kernel<<<ROWS, 256, 0, stream>>>(x_f32, ffn_ln_g + l * DIMC,
                                        ffn_ln_b + l * DIMC, nullptr, h_b);
    gemm_bf16_kernel<<<dim3(MLPC / 128, ROWS / 128), 256, 0, stream>>>(
        h_b, w1, ffn_b1 + l * MLPC, nullptr, nullptr, mlp_b, ROWS, MLPC, DIMC, 1);
    gemm_bf16_kernel<<<dim3(DIMC / 128, ROWS / 128), 256, 0, stream>>>(
        mlp_b, w2, ffn_b2 + l * DIMC, x_f32, x_f32, nullptr, ROWS, DIMC, MLPC, 0);
  }

  // final LN + classifier head
  ln_kernel<<<ROWS, 256, 0, stream>>>(x_f32, final_ln_g, final_ln_b,
                                      nullptr, h_b);
  proj_kernel<<<ROWS, 64, 0, stream>>>(h_b, proj_w, proj_b, (float*)d_out);
}
